// TopkDecoder_5351529251105
// MI455X (gfx1250) — matrix operand order repaired and checked by reading, not compiled, not run on hardware
//
#include <hip/hip_runtime.h>

typedef __attribute__((ext_vector_type(16))) __bf16 v16bf;
typedef __attribute__((ext_vector_type(8)))  float  v8f;
typedef unsigned short u16;

#define VOCAB   50257
#define DM      512
#define BSZ     32
#define ROWS    128          // B*K beams
#define TSTEPS  128
#define SEQW    132          // padded seq stride
#define SEQLEN  129
#define ENCL    128
#define BOSID   2
#define NT      3142         // ceil(50257/16) vocab tiles
#define KB      16           // 512/32 k-blocks
#define NWG     210          // gemm workgroups
#define TPW     15           // vocab tiles per workgroup (210*15 >= 3142)
#define NEGINF  (-3.0e38f)

__device__ __forceinline__ u16 f2bf(float f) {
  unsigned u = __float_as_uint(f);
  u += 0x7fffu + ((u >> 16) & 1u);      // round-to-nearest-even
  return (u16)(u >> 16);
}

// ---------------------------------------------------------------------------
// Pack Wo [512, VOCAB] f32 -> bf16 WMMA A-fragments.
// Layout: WoP[nt][kb][lane][i] (u16). A is 16(n) x 32(k) per tile.
// 16-bit A layout (matches B packing): lane<16 holds K 0..15 ; lane>=16 holds 16..31.
// A row M = lane%16 -> vocab column n = nt*16 + lane%16.
// ---------------------------------------------------------------------------
__global__ void k_pack_wo(const float* __restrict__ Wo, u16* __restrict__ WoP) {
  int nt = blockIdx.x;
  for (int idx = threadIdx.x; idx < 8192; idx += 256) {
    int kb   = idx >> 9;
    int lane = (idx >> 4) & 31;
    int i    = idx & 15;
    int n    = nt * 16 + (lane & 15);
    int kloc = i + ((lane >= 16) ? 16 : 0);
    int k    = kb * 32 + kloc;
    float v  = (n < VOCAB) ? Wo[(size_t)k * VOCAB + n] : 0.0f;
    WoP[(size_t)nt * 8192 + idx] = f2bf(v);
  }
}

// ---------------------------------------------------------------------------
// Masked-mean encoder context per batch: ctx[b][d]; also zero the fan-in ctr.
// ---------------------------------------------------------------------------
__global__ void k_ctx(const int* __restrict__ tok, const float* __restrict__ msk,
                      const float* __restrict__ E, float* __restrict__ ctx,
                      unsigned* __restrict__ counter) {
  int b = blockIdx.x;
  if (b == 0 && threadIdx.x == 0) *counter = 0u;
  float msum = 0.f;
  for (int i = 0; i < ENCL; ++i) msum += msk[b * ENCL + i];
  float inv = 1.0f / (msum + 1e-6f);
  for (int d = threadIdx.x; d < DM; d += blockDim.x) {
    float acc = 0.f;
    for (int i = 0; i < ENCL; ++i) {
      int t = tok[b * ENCL + i];
      acc += E[(size_t)t * DM + d] * msk[b * ENCL + i];
    }
    ctx[b * DM + d] = acc * inv;
  }
}

// ---------------------------------------------------------------------------
// One-shot step-0 h: h = E[BOS] + ctx, packed to bf16 B-fragments.
// B is 32(k) x 16(m): lane<16 -> beam col m0+lane, K 0..15; lane>=16 -> K 16..31.
// ---------------------------------------------------------------------------
__global__ void k_pack_h0(const float* __restrict__ E, const float* __restrict__ ctx,
                          float* __restrict__ hF, u16* __restrict__ hP) {
  __shared__ float hl[16 * DM];
  int w = blockIdx.x;
  for (int idx = threadIdx.x; idx < 16 * DM; idx += 256) {
    int r = w * 16 + (idx >> 9);
    int d = idx & 511;
    float h = E[(size_t)BOSID * DM + d] + ctx[(r >> 2) * DM + d];
    hl[idx] = h;
    hF[(size_t)r * DM + d] = h;
  }
  __syncthreads();
  for (int idx = threadIdx.x; idx < 8192; idx += 256) {
    int kb   = idx >> 9;
    int lane = (idx >> 4) & 31;
    int i    = idx & 15;
    int lrow = lane & 15;
    int k    = kb * 32 + ((lane < 16) ? 0 : 16) + i;
    hP[w * 8192 + idx] = f2bf(hl[lrow * DM + k]);
  }
}

// Branchless sorted-descending top-4 insert (v_cmp + v_cndmask, no exec churn).
// Update order tv3<-tv2<-tv1<-tv0 so each RHS reads pre-update values.
#define INS4(v, n)                                                            \
  do {                                                                        \
    float _v = (v); int _n = (n);                                             \
    bool c0 = _v > tv0, c1 = _v > tv1, c2 = _v > tv2, c3 = _v > tv3;          \
    tv3 = c3 ? (c2 ? tv2 : _v) : tv3;  ti3 = c3 ? (c2 ? ti2 : _n) : ti3;      \
    tv2 = c2 ? (c1 ? tv1 : _v) : tv2;  ti2 = c2 ? (c1 ? ti1 : _n) : ti2;      \
    tv1 = c1 ? (c0 ? tv0 : _v) : tv1;  ti1 = c1 ? (c0 ? ti0 : _n) : ti1;      \
    tv0 = c0 ? _v : tv0;               ti0 = c0 ? _n : ti0;                   \
  } while (0)

#define LD16(p) (*(const v16bf*)(p))

// ---------------------------------------------------------------------------
// One fused decode step:
//   phase 1 (all 210 WGs): WMMA GEMM (logits^T) + per-beam partial top-4.
//   phase 2 (last WG via fan-in): merge partials, softmax + beam select,
//   gather seqs, write probs, and produce next step's h + bf16 fragments.
// ---------------------------------------------------------------------------
__launch_bounds__(256)
__global__ void k_step(const u16* __restrict__ WoP, u16* hP,
                       float* pV, int* pI,
                       const float* __restrict__ E, const float* __restrict__ ctx,
                       const int* seqSrc, int* seqDst,
                       float* hF, float* probsOut, unsigned* counter,
                       int t, int first, int produceH) {
  int tid = threadIdx.x;
  int w = tid >> 5, lane = tid & 31;

  __shared__ float sv[256][4];
  __shared__ int   si[256][4];

  // ================= phase 1: GEMM + partial top-4 =================
  {
    v16bf bfr[KB];
    const u16* hb = hP + w * 8192 + lane * 16;
#pragma unroll
    for (int kb = 0; kb < KB; ++kb) bfr[kb] = LD16(hb + kb * 512);

    float tv0 = NEGINF, tv1 = NEGINF, tv2 = NEGINF, tv3 = NEGINF;
    int   ti0 = 0, ti1 = 0, ti2 = 0, ti3 = 0;

    int nt0   = (int)blockIdx.x * TPW;
    int ntEnd = nt0 + TPW;
    if (ntEnd > NT) ntEnd = NT;

    // depth-4 register ring keeps 4 loads in flight ahead of each wmma
    const u16* ap = WoP + (size_t)nt0 * 8192 + lane * 16;
    v16bf r0 = LD16(ap + 0 * 512);
    v16bf r1 = LD16(ap + 1 * 512);
    v16bf r2 = LD16(ap + 2 * 512);
    v16bf r3 = LD16(ap + 3 * 512);

    for (int nt = nt0; nt < ntEnd; ++nt) {
      v8f acc = {};
#pragma unroll
      for (int kb = 0; kb < KB; ++kb) {
        v16bf cur = r0; r0 = r1; r1 = r2; r2 = r3;
        // kb>=12 prefetches reach the next tile (WoP has 1 dummy pad tile)
        r3 = LD16(ap + (kb + 4) * 512);
        acc = __builtin_amdgcn_wmma_f32_16x16x32_bf16(
            false, cur, false, bfr[kb], (short)0, acc, false, false);
      }
      int nbase = nt * 16 + ((lane >= 16) ? 8 : 0);
#pragma unroll
      for (int j = 0; j < 8; ++j) {
        int n = nbase + j;
        float v = (n < VOCAB) ? acc[j] : NEGINF;
        INS4(v, n);
      }
      ap += 8192;
    }

    sv[tid][0] = tv0; sv[tid][1] = tv1; sv[tid][2] = tv2; sv[tid][3] = tv3;
    si[tid][0] = ti0; si[tid][1] = ti1; si[tid][2] = ti2; si[tid][3] = ti3;
    __syncthreads();
    if (lane < 16) {
#pragma unroll
      for (int j = 0; j < 4; ++j) INS4(sv[tid + 16][j], si[tid + 16][j]);
      int beam = w * 16 + lane;
      size_t o = ((size_t)blockIdx.x * ROWS + beam) * 4;
      pV[o + 0] = tv0; pV[o + 1] = tv1; pV[o + 2] = tv2; pV[o + 3] = tv3;
      pI[o + 0] = ti0; pI[o + 1] = ti1; pI[o + 2] = ti2; pI[o + 3] = ti3;
    }
  }

  // ================= fan-in: last WG does the serial tail =================
  __shared__ int sLast;
  __threadfence();
  __syncthreads();
  if (tid == 0) {
    unsigned old = atomicAdd(counter, 1u);
    sLast = (old == NWG - 1) ? 1 : 0;
  }
  __syncthreads();
  if (!sLast) return;
  if (tid == 0) *counter = 0u;   // reset for the next stream-ordered launch
  __threadfence();

  // ---- merge partials: 2 threads per beam row, then pairwise merge ----
  __shared__ float fv[ROWS][4];
  __shared__ int   fi[ROWS][4];
  {
    int row = tid >> 1, half = tid & 1;
    float tv0 = NEGINF, tv1 = NEGINF, tv2 = NEGINF, tv3 = NEGINF;
    int   ti0 = 0, ti1 = 0, ti2 = 0, ti3 = 0;
    for (int g = half; g < NWG; g += 2) {
      size_t o = ((size_t)g * ROWS + row) * 4;
      INS4(pV[o + 0], pI[o + 0]); INS4(pV[o + 1], pI[o + 1]);
      INS4(pV[o + 2], pI[o + 2]); INS4(pV[o + 3], pI[o + 3]);
    }
    sv[tid][0] = tv0; sv[tid][1] = tv1; sv[tid][2] = tv2; sv[tid][3] = tv3;
    si[tid][0] = ti0; si[tid][1] = ti1; si[tid][2] = ti2; si[tid][3] = ti3;
  }
  __syncthreads();
  if (tid < ROWS) {
    int row = tid;
    float tv0 = sv[2*row][0], tv1 = sv[2*row][1], tv2 = sv[2*row][2], tv3 = sv[2*row][3];
    int   ti0 = si[2*row][0], ti1 = si[2*row][1], ti2 = si[2*row][2], ti3 = si[2*row][3];
#pragma unroll
    for (int j = 0; j < 4; ++j) INS4(sv[2*row+1][j], si[2*row+1][j]);
    fv[row][0] = tv0; fv[row][1] = tv1; fv[row][2] = tv2; fv[row][3] = tv3;
    fi[row][0] = ti0; fi[row][1] = ti1; fi[row][2] = ti2; fi[row][3] = ti3;
  }
  __syncthreads();

  // ---- beam select per batch (softmax over top-4 raw logits; LSE cancels) ----
  __shared__ int parentS[ROWS], tokS[ROWS];
  if (tid < BSZ) {
    int b = tid;
    float p16[16];
    for (int k = 0; k < 4; ++k) {
      int r = b * 4 + k;
      float m = fv[r][0];
      float e0 = __expf(fv[r][0] - m), e1 = __expf(fv[r][1] - m);
      float e2 = __expf(fv[r][2] - m), e3 = __expf(fv[r][3] - m);
      float inv = 1.0f / (e0 + e1 + e2 + e3);
      p16[k*4+0] = e0*inv; p16[k*4+1] = e1*inv; p16[k*4+2] = e2*inv; p16[k*4+3] = e3*inv;
    }
    if (first) {
      // 4 h-rows of each batch identical (BOS + ctx); use its first row b*4, identity order
      for (int s = 0; s < 4; ++s) {
        parentS[b*4+s] = b*4;
        tokS[b*4+s]    = fi[b*4][s];
        probsOut[b*4+s] = p16[s];
      }
    } else {
      bool used[16];
      for (int c = 0; c < 16; ++c) used[c] = false;
      for (int s = 0; s < 4; ++s) {
        int best = 0; float bv = NEGINF;
        for (int c = 0; c < 16; ++c)
          if (!used[c] && p16[c] > bv) { bv = p16[c]; best = c; }
        used[best] = true;
        parentS[b*4+s] = b*4 + (best >> 2);
        tokS[b*4+s]    = fi[b*4 + (best >> 2)][best & 3];
        probsOut[b*4+s] = bv;
      }
    }
  }
  __syncthreads();

  // ---- sequence update ----
  if (first) {
    for (int idx = tid; idx < ROWS * SEQW; idx += 256) {
      int r = idx / SEQW, c = idx - r * SEQW;
      int v = (c == 0) ? BOSID : ((c == 1) ? tokS[r] : 0);
      seqDst[idx] = v;
    }
  } else {
    for (int idx = tid; idx < ROWS * SEQW; idx += 256) {
      int r = idx / SEQW, c = idx - r * SEQW;
      seqDst[idx] = seqSrc[parentS[r] * SEQW + c];
    }
    __syncthreads();
    if (tid < ROWS) seqDst[tid * SEQW + (t + 1)] = tokS[tid];
  }

  // ---- produce next step's h (f32) + bf16 fragments ----
  if (produceH) {
    __syncthreads();
    for (int idx = tid; idx < ROWS * DM; idx += 256) {
      int r = idx >> 9, d = idx & 511;
      hF[idx] = E[(size_t)tokS[r] * DM + d] + ctx[(r >> 2) * DM + d];
    }
    __threadfence_block();
    __syncthreads();
    for (int idx = tid; idx < 8 * 8192; idx += 256) {
      int w2    = idx >> 13;
      int kb    = (idx >> 9) & 15;
      int lane2 = (idx >> 4) & 31;
      int i     = idx & 15;
      int lrow  = lane2 & 15;
      int k     = kb * 32 + ((lane2 < 16) ? 0 : 16) + i;
      hP[idx] = f2bf(hF[(w2 * 16 + lrow) * DM + k]);
    }
  }
}

// ---------------------------------------------------------------------------
// Final output assembly: seqs (as float), then probs (already written), then h
// ---------------------------------------------------------------------------
__global__ void k_fin(const int* __restrict__ seq, const float* __restrict__ hF,
                      float* __restrict__ out) {
  int i = blockIdx.x * 256 + threadIdx.x;
  if (i < ROWS * SEQLEN) {
    out[i] = (float)seq[(i / SEQLEN) * SEQW + (i % SEQLEN)];
  } else {
    int k = i - (ROWS * SEQLEN + ROWS);
    if (k >= 0 && k < ROWS * DM) out[ROWS * SEQLEN + ROWS + k] = hF[k];
  }
}

extern "C" void kernel_launch(void* const* d_in, const int* in_sizes, int n_in,
                              void* d_out, int out_size, void* d_ws, size_t ws_size,
                              hipStream_t stream) {
  (void)in_sizes; (void)n_in; (void)out_size; (void)ws_size;
  const int*   tok = (const int*)d_in[0];
  const float* msk = (const float*)d_in[1];
  const float* E   = (const float*)d_in[2];
  const float* Wo  = (const float*)d_in[3];

  char* ws = (char*)d_ws;
  size_t o = 0;
  auto alloc = [&](size_t bytes) -> void* {
    void* p = ws + o;
    o = (o + bytes + 255) & ~(size_t)255;
    return p;
  };
  // +1 dummy tile so the pipelined tail prefetch stays in-bounds
  u16*   WoP   = (u16*)  alloc((size_t)(NT + 1) * 8192 * 2);
  u16*   hP    = (u16*)  alloc((size_t)8 * 8192 * 2);      // per-step B fragments
  float* hF    = (float*)alloc((size_t)ROWS * DM * 4);     // f32 h (output)
  float* ctx   = (float*)alloc((size_t)BSZ * DM * 4);
  float* pV    = (float*)alloc((size_t)NWG * ROWS * 4 * 4);
  int*   pI    = (int*)  alloc((size_t)NWG * ROWS * 4 * 4);
  int*   seqA  = (int*)  alloc((size_t)ROWS * SEQW * 4);
  int*   seqB  = (int*)  alloc((size_t)ROWS * SEQW * 4);
  unsigned* counter = (unsigned*)alloc(256);
  float* probsOut = (float*)d_out + ROWS * SEQLEN;

  k_pack_wo<<<NT, 256, 0, stream>>>(Wo, WoP);
  k_ctx<<<BSZ, 256, 0, stream>>>(tok, msk, E, ctx, counter);
  k_pack_h0<<<8, 256, 0, stream>>>(E, ctx, hF, hP);

  // step 0 (all beams seeded with BOS)
  k_step<<<NWG, 256, 0, stream>>>(WoP, hP, pV, pI, E, ctx,
                                  seqA, seqA, hF, probsOut, counter,
                                  0, /*first=*/1, /*produceH=*/1);
  int* cur = seqA; int* nxt = seqB;
  for (int t = 1; t < TSTEPS; ++t) {
    k_step<<<NWG, 256, 0, stream>>>(WoP, hP, pV, pI, E, ctx,
                                    cur, nxt, hF, probsOut, counter,
                                    t, /*first=*/0, /*produceH=*/(t < TSTEPS - 1) ? 1 : 0);
    int* tmp = cur; cur = nxt; nxt = tmp;
  }

  int total = ROWS * SEQLEN + ROWS + ROWS * DM;
  k_fin<<<(total + 255) / 256, 256, 0, stream>>>(cur, hF, (float*)d_out);
}
